// SCGLA_24034636989267
// MI455X (gfx1250) — compile-verified, matrix operand order checked by reading, not run on hardware
//
#include <hip/hip_runtime.h>
#include <hip/hip_bf16.h>

// ---------------------------------------------------------------------------
// Problem constants (from reference): N=2, Cch=64, H=W=96
//   HW = 9216, L = 2*HW = 18432, N_HASHES=4, CHUNK=144, buckets/hash = 128
//   Q = 4*L = 73728 sorted positions per batch. C (match) = 16, C*RED = 64.
// ---------------------------------------------------------------------------
#define HW_   9216
#define L_    18432
#define Q_    73728
#define CH_   144
#define NB_   128

typedef __attribute__((ext_vector_type(16))) _Float16 v16h;
typedef __attribute__((ext_vector_type(8)))  float    v8f;

// ISA 7.12.2 K-index map for 16-bit A/B fragments of v_wmma_*_16x16x32_f16:
//   K = (h&7) + ((h>>3)<<4) + ((lane>>4)<<3)
// => per lane, halves 0..7 and 8..15 are two contiguous 8-half K-groups.
__device__ __forceinline__ int wmma_kmap(int lane, int h) {
  return (h & 7) | ((h & 8) << 1) | ((lane >> 4) << 3);
}

// Load a fragment stored fragment-ready (lane*16 contiguous halves): 2x b128.
__device__ __forceinline__ v16h load_frag(const _Float16* p) {
  v16h r;
  ((uint4*)&r)[0] = ((const uint4*)p)[0];
  ((uint4*)&r)[1] = ((const uint4*)p)[1];
  return r;
}
// Load an A fragment from row-major memory: two contiguous 8-half groups at
// p (k=base..base+7) and p+16 (k=base+16..base+23): 2x b128.
__device__ __forceinline__ v16h load_a_split(const _Float16* p) {
  v16h r;
  ((uint4*)&r)[0] = *(const uint4*)p;
  ((uint4*)&r)[1] = *(const uint4*)(p + 16);
  return r;
}

#define WMMA_F32_F16(a, b, c) \
  __builtin_amdgcn_wmma_f32_16x16x32_f16(false, (a), false, (b), (short)0, (c), false, false)

// ---------------------------------------------------------------------------
// K1: WMMA implicit-GEMM fused 3x3 convs (match16|assembly64|fc64 = 144 ch).
//   out(pixels x 144) = patches(pixels x 576) @ Wt(576 x 144)
// K is REORDERED as k' = (ky*3+kx)*64 + ci so that, with the input tile staged
// channel-minor with x-halo ([6 rows][98 cols][64 ch], halo pre-zeroed), every
// A fragment is two contiguous ds_load_b128 with no per-half address math.
// One workgroup per (b, src, 4-row block); 24 M-tiles of 16 pixels, 8 waves.
// ---------------------------------------------------------------------------
__global__ __launch_bounds__(256) void conv_wmma_kernel(
    const float* __restrict__ in1, const float* __restrict__ in2,
    const float* __restrict__ wm, const float* __restrict__ wa,
    const float* __restrict__ wf, float* __restrict__ embedAll) {
  extern __shared__ char smem_raw[];
  _Float16* sW  = (_Float16*)smem_raw;   // 18*9*512 = 82944 halves (frag-ready)
  _Float16* sIn = sW + 18 * 9 * 512;     // 6*98*64  = 37632 halves
  const int tid = threadIdx.x, lane = tid & 31, wid = tid >> 5;
  const int blk = blockIdx.x;            // [N*2*24)
  const int rb = blk % 24;
  const int src = (blk / 24) % 2;
  const int b = blk / 48;
  const int y0 = rb * 4;
  const float* inp = (src == 0 ? in1 : in2) + (size_t)b * 64 * HW_;

  // Weights fragment-ready with reordered K: flat index = ci*9 + (ky*3+kx)
  for (int idx = tid; idx < 18 * 9 * 512; idx += 256) {
    const int tile = idx >> 9, rem = idx & 511, l = rem >> 4, hh = rem & 15;
    const int kk = (tile / 9) * 32 + wmma_kmap(l, hh);  // k' in 0..575
    const int c = (tile % 9) * 16 + (l & 15);
    const float* w; int co;
    if (c < 16)      { w = wm; co = c; }
    else if (c < 80) { w = wa; co = c - 16; }
    else             { w = wf; co = c - 80; }
    sW[idx] = (_Float16)w[(size_t)co * 576 + (kk & 63) * 9 + (kk >> 6)];
  }
  // Input tile channel-minor, rows y0-1..y0+4, cols -1..96 (halo zeroed)
  for (int idx = tid; idx < 6 * 98 * 64; idx += 256) {
    const int ci = idx & 63;
    const int xx = (idx >> 6) % 98;   // global x = xx-1
    const int ry = (idx >> 6) / 98;   // global y = y0-1+ry
    const int yy = y0 - 1 + ry;
    const int xg = xx - 1;
    sIn[idx] = (yy >= 0 && yy < 96 && xg >= 0 && xg < 96)
                   ? (_Float16)inp[(size_t)ci * HW_ + yy * 96 + xg]
                   : (_Float16)0.f;
  }
  __syncthreads();

  const v8f vzero = {0.f, 0.f, 0.f, 0.f, 0.f, 0.f, 0.f, 0.f};
  const int m16 = lane & 15;
  const int basek = (lane >> 4) << 3;
  for (int mt = wid; mt < 24; mt += 8) {  // 3 M-tiles per wave
    const int yloc = mt / 6;
    const int xt = (mt % 6) * 16;
    const int x = xt + m16;
    v8f acc[9];
#pragma unroll
    for (int n = 0; n < 9; ++n) acc[n] = vzero;
#pragma unroll
    for (int kc = 0; kc < 18; ++kc) {
      const int g = kc >> 1;               // compile-time after unroll
      const int ky = g / 3, kx = g % 3;
      const int ci0 = ((kc & 1) << 5) + basek;
      const v16h a =
          load_a_split(sIn + ((yloc + ky) * 98 + x + kx) * 64 + ci0);
#pragma unroll
      for (int nt = 0; nt < 9; ++nt)
        acc[nt] = WMMA_F32_F16(a, load_frag(sW + (kc * 9 + nt) * 512 + lane * 16), acc[nt]);
    }
#pragma unroll
    for (int nt = 0; nt < 9; ++nt) {
#pragma unroll
      for (int v = 0; v < 8; ++v) {
        const int mm = v + basek;
        const int t = src * HW_ + (y0 + yloc) * 96 + xt + mm;
        embedAll[((size_t)b * L_ + t) * 144 + nt * 16 + m16] = acc[nt][v];
      }
    }
  }
}

// ---------------------------------------------------------------------------
// K2: LSH hash: argmax over 128 per hash of x_embed(16) @ rot(16,512).
// ---------------------------------------------------------------------------
__global__ __launch_bounds__(256) void hash_kernel(
    const float* __restrict__ embedAll, const float* __restrict__ rot,
    int* __restrict__ keys) {
  const int idx = blockIdx.x * 256 + threadIdx.x;
  if (idx >= 2 * L_) return;
  const int b = idx / L_, t = idx % L_;
  float xv[16];
  const float* e = embedAll + ((size_t)b * L_ + t) * 144;
#pragma unroll
  for (int c = 0; c < 16; ++c) xv[c] = e[c];
  for (int hh = 0; hh < 4; ++hh) {
    float best = -1e30f;
    int bi = 0;
    for (int i = 0; i < 128; ++i) {
      float s = 0.f;
#pragma unroll
      for (int c = 0; c < 16; ++c) s += xv[c] * rot[c * 512 + hh * 128 + i];
      if (s > best) { best = s; bi = i; }  // first max == argmax
    }
    keys[(size_t)b * Q_ + hh * L_ + t] = bi + hh * 128;
  }
}

// ---------------------------------------------------------------------------
// Stable counting sort (matches stable argsort). counts/offsets: [b][bin][blk]
// ---------------------------------------------------------------------------
__global__ void sort_hist_kernel(const int* __restrict__ keys,
                                 int* __restrict__ counts) {
  __shared__ int hist[512];
  const int b = blockIdx.x / 144, blk = blockIdx.x % 144;
  for (int i = threadIdx.x; i < 512; i += blockDim.x) hist[i] = 0;
  __syncthreads();
  const int key = keys[(size_t)b * Q_ + blk * 512 + threadIdx.x];
  atomicAdd(&hist[key], 1);
  __syncthreads();
  for (int i = threadIdx.x; i < 512; i += blockDim.x)
    counts[((size_t)b * 512 + i) * 144 + blk] = hist[i];
}

__global__ void sort_scan_kernel(const int* __restrict__ counts,
                                 int* __restrict__ offsets) {
  __shared__ int totals[512];
  __shared__ int binStart[512];
  const int b = blockIdx.x, bin = threadIdx.x;
  int sum = 0;
  for (int blk = 0; blk < 144; ++blk)
    sum += counts[((size_t)b * 512 + bin) * 144 + blk];
  totals[bin] = sum;
  __syncthreads();
  if (bin == 0) {
    int r = 0;
    for (int i = 0; i < 512; ++i) { binStart[i] = r; r += totals[i]; }
  }
  __syncthreads();
  int run = binStart[bin];
  for (int blk = 0; blk < 144; ++blk) {
    offsets[((size_t)b * 512 + bin) * 144 + blk] = run;
    run += counts[((size_t)b * 512 + bin) * 144 + blk];
  }
}

__global__ void sort_scatter_kernel(const int* __restrict__ keys,
                                    const int* __restrict__ offsets,
                                    int* __restrict__ sortedIdx,
                                    int* __restrict__ undo) {
  __shared__ int skey[512];
  const int b = blockIdx.x / 144, blk = blockIdx.x % 144;
  const int t = threadIdx.x;
  const int g = blk * 512 + t;
  const int key = keys[(size_t)b * Q_ + g];
  skey[t] = key;
  __syncthreads();
  int rank = 0;
  for (int j = 0; j < t; ++j) rank += (skey[j] == key) ? 1 : 0;  // stable
  const int pos = offsets[((size_t)b * 512 + key) * 144 + blk] + rank;
  sortedIdx[(size_t)b * Q_ + pos] = g;
  undo[(size_t)b * Q_ + g] = pos;
}

// ---------------------------------------------------------------------------
// K6a: gather sorted fp16 rows: gath[b][q][0:16)=x, [16:80)=y, [80:144)=fc
// K6b: transposed y gather: ybT[b][e][q] (so yb fragments stage with b128s)
// ---------------------------------------------------------------------------
__global__ __launch_bounds__(256) void gather_kernel(
    const float* __restrict__ embedAll, const int* __restrict__ sortedIdx,
    _Float16* __restrict__ gath) {
  const long long idx = (long long)blockIdx.x * 256 + threadIdx.x;
  if (idx >= (long long)2 * Q_ * 144) return;
  const int c = (int)(idx % 144);
  const long long q = idx / 144;
  const int b = (int)(q / Q_);
  const int qq = (int)(q % Q_);
  const int row = sortedIdx[(size_t)b * Q_ + qq] % L_;
  gath[idx] = (_Float16)embedAll[((size_t)b * L_ + row) * 144 + c];
}

__global__ __launch_bounds__(256) void gatherT_kernel(
    const float* __restrict__ embedAll, const int* __restrict__ sortedIdx,
    _Float16* __restrict__ ybT) {
  const long long idx = (long long)blockIdx.x * 256 + threadIdx.x;
  if (idx >= (long long)2 * 64 * Q_) return;
  const int q = (int)(idx % Q_);
  const int e = (int)((idx / Q_) % 64);
  const int b = (int)(idx / ((long long)64 * Q_));
  const int row = sortedIdx[(size_t)b * Q_ + q] % L_;
  ybT[idx] = (_Float16)embedAll[((size_t)b * L_ + row) * 144 + 16 + e];
}

// ---------------------------------------------------------------------------
// K7: bucketed LSH attention. One workgroup (16 waves, 512 threads) per bucket
// (1024 total).  LDS (~286 KB) pins 1 WG per WGP, so 16 waves give 4 waves per
// SIMD32 for latency hiding.
//   rawC[i][j] (col-major, 144x448) = x_match@xbT + relu(fb@W1+b1)@W2 + b2
//   column softmax over j, ret(144x64) = score @ yb_adj.  All GEMMs WMMA f16.
// B operands pre-swizzled fragment-ready (2x ds_load_b128 per fragment);
// fb/x_match A fragments loaded directly from global as 2 contiguous b128s;
// rawC written/swept with packed b128s; next strip prefetched.
// ---------------------------------------------------------------------------
__global__ __launch_bounds__(512) void lsh_attn_kernel(
    const _Float16* __restrict__ gath, const _Float16* __restrict__ ybT,
    const float* __restrict__ fcw1, const float* __restrict__ fcb1,
    const float* __restrict__ fcw2, const float* __restrict__ fcb2,
    float* __restrict__ retS, float* __restrict__ bscS) {
  extern __shared__ char smem_raw[];
  _Float16* rawC  = (_Float16*)smem_raw;   // 144*448 col-major scores
  _Float16* sW1f  = rawC + 144 * 448;      // 18 tiles * 512
  _Float16* sW2f  = sW1f + 18 * 512;       // 45 tiles * 512 (K-pad 0)
  _Float16* sXbTf = sW2f + 45 * 512;       //  9 tiles * 512 (K-pad 0)
  float* sB1  = (float*)(sXbTf + 9 * 512); // 144
  float* sB2  = sB1 + 144;                 // 144
  float* sInvZ = sB2 + 144;                // 144
  _Float16* sPool = (_Float16*)(sInvZ + 144);  // 40960 halves: h1f / sYbF

  const int tid = threadIdx.x;
  const int lane = tid & 31;
  const int wid = tid >> 5;                // 0..15
  const int blk = blockIdx.x;
  const int b = blk >> 9;
  const int hsh = (blk >> 7) & 3;
  const int k = blk & 127;
  const size_t rowBase = (size_t)b * Q_;
  const int q0  = hsh * L_ + k * CH_;
  const int q0m = hsh * L_ + ((k + NB_ - 1) & 127) * CH_;
  const int q0p = hsh * L_ + ((k + 1) & 127) * CH_;
  const int m16 = lane & 15;
  const int basek = (lane >> 4) << 3;
  const v8f vzero = {0.f, 0.f, 0.f, 0.f, 0.f, 0.f, 0.f, 0.f};

  // ---- Phase 0: fragment-swizzle weights + xbT ----
  for (int idx = tid; idx < 18 * 512; idx += 512) {  // W1: K=64
    const int tile = idx >> 9, rem = idx & 511, l = rem >> 4, hh = rem & 15;
    const int kk = (tile / 9) * 32 + wmma_kmap(l, hh);
    const int col = (tile % 9) * 16 + (l & 15);
    sW1f[idx] = (_Float16)fcw1[kk * 144 + col];
  }
  for (int idx = tid; idx < 45 * 512; idx += 512) {  // W2: K=144 (pad 160)
    const int tile = idx >> 9, rem = idx & 511, l = rem >> 4, hh = rem & 15;
    const int kk = (tile / 9) * 32 + wmma_kmap(l, hh);
    const int col = (tile % 9) * 16 + (l & 15);
    sW2f[idx] = (kk < 144) ? (_Float16)fcw2[kk * 144 + col] : (_Float16)0.f;
  }
  for (int i = tid; i < 144; i += 512) { sB1[i] = fcb1[i]; sB2[i] = fcb2[i]; }
  for (int slot = tid; slot < 9 * 32; slot += 512) {  // xbT: K=16 (pad 32)
    const int nt = slot >> 5, l = slot & 31;
    const int i = nt * 16 + (l & 15);
    const int bk = (l >> 4) << 3;
    uint4* dst = (uint4*)(sXbTf + nt * 512 + l * 16);
    dst[0] = *(const uint4*)(gath + (rowBase + q0 + i) * 144 + bk);
    dst[1] = make_uint4(0, 0, 0, 0);
  }
  __syncthreads();

  // ---- Phase 1: each wave computes 16-row j-strips of rawC ----
  _Float16* h1f = sPool + wid * 2560;  // 5 kc * 512, A-fragment order
  for (int s = wid; s < 27; s += 16) {
    const int sub = s / 9;  // 0:self 1:prev 2:next (strips never straddle)
    const int qb = (sub == 0) ? q0 : (sub == 1) ? q0m : q0p;
    const int r0 = (s % 9) * 16;
    const _Float16* growM = gath + (rowBase + qb + r0 + m16) * 144;

    // prefetch next strip's rows (global_prefetch_b8)
    const int sn = s + 16;
    if (sn < 27) {
      const int subn = sn / 9;
      const int qbn = (subn == 0) ? q0 : (subn == 1) ? q0m : q0p;
      __builtin_prefetch(gath + (rowBase + qbn + (sn % 9) * 16 + m16) * 144, 0, 1);
    }

    // h1 = relu(fb @ W1 + b1): fb A-fragments straight from global (ch 80+)
    v8f acc[9];
#pragma unroll
    for (int n = 0; n < 9; ++n) acc[n] = vzero;
#pragma unroll
    for (int kc = 0; kc < 2; ++kc) {
      const v16h a = load_a_split(growM + 80 + kc * 32 + basek);
#pragma unroll
      for (int nt = 0; nt < 9; ++nt)
        acc[nt] = WMMA_F32_F16(a, load_frag(sW1f + (kc * 9 + nt) * 512 + lane * 16), acc[nt]);
    }
    // relu+bias -> h1f in A-fragment order (scatter once, vector-load 5x)
#pragma unroll
    for (int nt = 0; nt < 9; ++nt) {
      const int col = nt * 16 + m16;
      const int kc2 = col >> 5, kl = col & 31;
      const int lphi = (kl >> 3) & 1;
      const int hh = (kl & 7) | ((kl >> 4) << 3);
      const float bias = sB1[col];
#pragma unroll
      for (int v = 0; v < 8; ++v) {
        const int m = v + basek;
        h1f[kc2 * 512 + (m | (lphi << 4)) * 16 + hh] =
            (_Float16)fmaxf(acc[nt][v] + bias, 0.f);
      }
    }
    *(uint4*)(h1f + 4 * 512 + lane * 16 + 8) = make_uint4(0, 0, 0, 0);  // K-pad

    // x_match A fragment: registers only (normalize over 16 channels)
    v16h ax;
    {
      _Float16 xl[8], xh[8];
      *(uint4*)xl = *(const uint4*)growM;        // ch 0..7
      *(uint4*)xh = *(const uint4*)(growM + 8);  // ch 8..15
      float ss = 0.f;
#pragma unroll
      for (int c = 0; c < 8; ++c) {
        const float a1 = (float)xl[c], a2 = (float)xh[c];
        ss += a1 * a1 + a2 * a2;
      }
      const float inv = 1.f / fmaxf(sqrtf(ss), 5e-5f);
      const bool lowk = (lane < 16);
#pragma unroll
      for (int hh = 0; hh < 8; ++hh) {
        const float xv = lowk ? (float)xl[hh] : (float)xh[hh];
        ax[hh] = (_Float16)(xv * inv);
        ax[hh + 8] = (_Float16)0.f;  // K 16..31 zero-pad
      }
    }

    // rawC strip = xm@xbT + h1@W2 (+ b2)
#pragma unroll
    for (int n = 0; n < 9; ++n) acc[n] = vzero;
#pragma unroll
    for (int nt = 0; nt < 9; ++nt)
      acc[nt] = WMMA_F32_F16(ax, load_frag(sXbTf + nt * 512 + lane * 16), acc[nt]);
#pragma unroll
    for (int kc = 0; kc < 5; ++kc) {
      const v16h a = load_frag(h1f + kc * 512 + lane * 16);
#pragma unroll
      for (int nt = 0; nt < 9; ++nt)
        acc[nt] = WMMA_F32_F16(a, load_frag(sW2f + (kc * 9 + nt) * 512 + lane * 16), acc[nt]);
    }
    // packed store: 8 consecutive j at fixed i -> one b128 per tile
#pragma unroll
    for (int nt = 0; nt < 9; ++nt) {
      const int i = nt * 16 + m16;
      const float bias = sB2[i];
      _Float16 pk[8];
#pragma unroll
      for (int v = 0; v < 8; ++v) pk[v] = (_Float16)(acc[nt][v] + bias);
      *(uint4*)(rawC + i * 448 + s * 16 + basek) = *(const uint4*)pk;
    }
  }
  __syncthreads();

  // ---- Phase 2: column softmax over j (vectorized b128 sweeps) ----
  if (tid < 144) {
    _Float16* rowp = rawC + tid * 448;
    float m = -1e30f;
    for (int j8 = 0; j8 < 54; ++j8) {  // 432 = 54*8
      _Float16 buf[8];
      *(uint4*)buf = ((const uint4*)rowp)[j8];
#pragma unroll
      for (int c = 0; c < 8; ++c) m = fmaxf(m, (float)buf[c]);
    }
    float ssum = 0.f;
    for (int j8 = 0; j8 < 54; ++j8) {
      _Float16 buf[8];
      *(uint4*)buf = ((const uint4*)rowp)[j8];
#pragma unroll
      for (int c = 0; c < 8; ++c) {
        const float e = __expf((float)buf[c] - m);
        ssum += e;
        buf[c] = (_Float16)e;
      }
      ((uint4*)rowp)[j8] = *(const uint4*)buf;
    }
    ((uint4*)rowp)[54] = make_uint4(0, 0, 0, 0);  // K-pad score = 0
    ((uint4*)rowp)[55] = make_uint4(0, 0, 0, 0);
    sInvZ[tid] = 1.f / ssum;
    bscS[rowBase + q0 + tid] = m + __logf(ssum);  // logsumexp
  }
  // Stage yb fragments from transposed ybT (16B vector copies)
  _Float16* sYbF = sPool;  // 56 tiles (14 kc x 4 et) * 512
  for (int slot = tid; slot < 56 * 32; slot += 512) {
    const int kc = slot >> 7;
    const int et = (slot >> 5) & 3;
    const int l = slot & 31;
    const int e = et * 16 + (l & 15);
    const int bk = (l >> 4) << 3;
    const _Float16* ybase = ybT + ((size_t)b * 64 + e) * Q_;
    uint4* dst = (uint4*)(sYbF + (kc * 4 + et) * 512 + l * 16);
#pragma unroll
    for (int g = 0; g < 2; ++g) {
      const int j0 = kc * 32 + g * 16 + bk;  // 8-aligned groups never straddle
      if (j0 < 432) {
        const int sub = j0 / 144;
        const int qb = (sub == 0) ? q0 : (sub == 1) ? q0m : q0p;
        dst[g] = *(const uint4*)(ybase + qb + (j0 % 144));
      } else {
        dst[g] = make_uint4(0, 0, 0, 0);
      }
    }
  }
  __syncthreads();

  // ---- Phase 3: ret = score(144x448) @ yb(448x64), scale by 1/Z ----
  for (int tile = wid; tile < 36; tile += 16) {
    const int it = tile >> 2, et = tile & 3;
    const _Float16* arow = rawC + (it * 16 + m16) * 448 + basek;
    v8f acc = vzero;
#pragma unroll
    for (int kc = 0; kc < 14; ++kc)
      acc = WMMA_F32_F16(load_a_split(arow + kc * 32),
                         load_frag(sYbF + (kc * 4 + et) * 512 + lane * 16), acc);
#pragma unroll
    for (int v = 0; v < 8; ++v) {
      const int i = it * 16 + v + basek;
      retS[(rowBase + q0 + i) * 64 + et * 16 + m16] = acc[v] * sInvZ[i];
    }
  }
}

// ---------------------------------------------------------------------------
// K8: unsort + cross-hash softmax combine + residual, write out1|out2.
// ---------------------------------------------------------------------------
__global__ __launch_bounds__(256) void combine_kernel(
    const float* __restrict__ retS, const float* __restrict__ bscS,
    const int* __restrict__ undo, const float* __restrict__ in1,
    const float* __restrict__ in2, float* __restrict__ out) {
  const long long idx = (long long)blockIdx.x * 256 + threadIdx.x;
  if (idx >= (long long)2 * L_ * 64) return;
  const int e = (int)(idx % 64);
  const long long p = idx / 64;
  const int t = (int)(p % L_);
  const int b = (int)(p / L_);
  float sc[4], vv[4], m = -1e30f;
#pragma unroll
  for (int hh = 0; hh < 4; ++hh) {
    const int pos = undo[(size_t)b * Q_ + hh * L_ + t];
    sc[hh] = bscS[(size_t)b * Q_ + pos];
    vv[hh] = retS[((size_t)b * Q_ + pos) * 64 + e];
    m = fmaxf(m, sc[hh]);
  }
  float Z = 0.f, r = 0.f;
#pragma unroll
  for (int hh = 0; hh < 4; ++hh) {
    const float pw = __expf(sc[hh] - m);
    Z += pw;
    r += pw * vv[hh];
  }
  r /= Z;
  const int src = t / HW_, pix = t % HW_;
  const float* inp = (src == 0) ? in1 : in2;
  out[(size_t)src * (2 * 64 * HW_) + ((size_t)b * 64 + e) * HW_ + pix] =
      r + inp[((size_t)b * 64 + e) * HW_ + pix];
}

// ---------------------------------------------------------------------------
// Launch
// ---------------------------------------------------------------------------
extern "C" void kernel_launch(void* const* d_in, const int* in_sizes, int n_in,
                              void* d_out, int out_size, void* d_ws,
                              size_t ws_size, hipStream_t stream) {
  const float* in1 = (const float*)d_in[0];
  const float* in2 = (const float*)d_in[1];
  const float* wm  = (const float*)d_in[2];
  const float* wa  = (const float*)d_in[3];
  const float* wf  = (const float*)d_in[4];
  const float* fw1 = (const float*)d_in[5];
  const float* fb1 = (const float*)d_in[6];
  const float* fw2 = (const float*)d_in[7];
  const float* fb2 = (const float*)d_in[8];
  const float* rot = (const float*)d_in[9];

  char* ws = (char*)d_ws;
  size_t off = 0;
  auto alloc = [&](size_t bytes) {
    void* p = ws + off;
    off += (bytes + 255) & ~(size_t)255;
    return p;
  };
  float*    embedAll  = (float*)alloc((size_t)2 * L_ * 144 * 4);   // 21.2 MB
  int*      keys      = (int*)alloc((size_t)2 * Q_ * 4);
  int*      counts    = (int*)alloc((size_t)2 * 512 * 144 * 4);
  int*      offsets   = (int*)alloc((size_t)2 * 512 * 144 * 4);
  int*      sortedIdx = (int*)alloc((size_t)2 * Q_ * 4);
  int*      undo      = (int*)alloc((size_t)2 * Q_ * 4);
  _Float16* gath      = (_Float16*)alloc((size_t)2 * Q_ * 144 * 2); // 42.5 MB
  _Float16* ybT       = (_Float16*)alloc((size_t)2 * 64 * Q_ * 2);  // 18.9 MB
  float*    retS      = (float*)alloc((size_t)2 * Q_ * 64 * 4);     // 37.7 MB
  float*    bscS      = (float*)alloc((size_t)2 * Q_ * 4);
  (void)ws_size; (void)in_sizes; (void)n_in; (void)out_size;

  // K1: WMMA implicit-GEMM convs (reordered-K, channel-minor input tile)
  constexpr size_t CONV_SMEM = (size_t)(18 * 9 * 512 + 6 * 98 * 64) * 2;  // 241152
  conv_wmma_kernel<<<2 * 2 * 24, 256, CONV_SMEM, stream>>>(in1, in2, wm, wa, wf,
                                                           embedAll);
  // K2: LSH hash codes
  hash_kernel<<<(2 * L_ + 255) / 256, 256, 0, stream>>>(embedAll, rot, keys);
  // K3..K5: stable counting sort
  sort_hist_kernel<<<2 * 144, 512, 0, stream>>>(keys, counts);
  sort_scan_kernel<<<2, 512, 0, stream>>>(counts, offsets);
  sort_scatter_kernel<<<2 * 144, 512, 0, stream>>>(keys, offsets, sortedIdx, undo);
  // K6: gathers (row-major f16 + transposed y)
  gather_kernel<<<(int)(((long long)2 * Q_ * 144 + 255) / 256), 256, 0, stream>>>(
      embedAll, sortedIdx, gath);
  gatherT_kernel<<<(int)(((long long)2 * 64 * Q_ + 255) / 256), 256, 0, stream>>>(
      embedAll, sortedIdx, ybT);
  // K7: WMMA bucket attention, 1024 buckets, 16 waves each, ~286 KB LDS
  constexpr size_t ATTN_SMEM =
      (size_t)(144 * 448 + 18 * 512 + 45 * 512 + 9 * 512 + 80 * 512) * 2 +
      (size_t)3 * 144 * 4;  // 286400
  lsh_attn_kernel<<<2 * 4 * NB_, 512, ATTN_SMEM, stream>>>(gath, ybT, fw1, fb1,
                                                           fw2, fb2, retS, bscS);
  // K8: unsort + hash-softmax combine + residual
  combine_kernel<<<(int)(((long long)2 * L_ * 64 + 255) / 256), 256, 0, stream>>>(
      retS, bscS, undo, in1, in2, (float*)d_out);
}
